// CurveBackBone_49563922596245
// MI455X (gfx1250) — matrix operand order repaired and checked by reading, not compiled, not run on hardware
//
#include <hip/hip_runtime.h>
#include <hip/hip_bf16.h>
#include <cstdint>

// ---------------------------------------------------------------- constants
#define NPTS 262144      // B * N_PER_B
#define SEG  65536       // points per batch
#define GS   64          // GROUP_SIZE
#define HORD 9           // Hilbert order

typedef __attribute__((ext_vector_type(16))) __bf16 bf16x16;
typedef __attribute__((ext_vector_type(8)))  float  floatx8;

// ---------------------------------------------------------------- hilbert
__device__ inline int hilbert3(int X0, int X1, int X2) {
    const int M = 1 << (HORD - 1);
    for (int Q = M; Q > 1; Q >>= 1) {
        const int P = Q - 1;
        // i = 0
        if (X0 & Q) X0 ^= P;
        // i = 1
        if (X1 & Q) { X0 ^= P; }
        else        { int t = (X0 ^ X1) & P; X0 ^= t; X1 ^= t; }
        // i = 2
        if (X2 & Q) { X0 ^= P; }
        else        { int t = (X0 ^ X2) & P; X0 ^= t; X2 ^= t; }
    }
    X1 ^= X0;
    X2 ^= X1;
    int t = 0;
    for (int Q = M; Q > 1; Q >>= 1) if (X2 & Q) t ^= (Q - 1);
    X0 ^= t; X1 ^= t; X2 ^= t;
    int code = 0;
    for (int b = HORD - 1; b >= 0; --b) {
        code = (code << 1) | ((X0 >> b) & 1);
        code = (code << 1) | ((X1 >> b) & 1);
        code = (code << 1) | ((X2 >> b) & 1);
    }
    return code;
}

__global__ void build_keys_kernel(const int* __restrict__ coors,
                                  unsigned long long* __restrict__ k1,
                                  unsigned long long* __restrict__ k2, int n) {
    int i = blockIdx.x * blockDim.x + threadIdx.x;
    if (i >= n) return;
    int z = coors[i * 4 + 1], y = coors[i * 4 + 2], x = coors[i * 4 + 3];
    int c1 = hilbert3(z, y, x);   // coors[:,1:4]
    int c2 = hilbert3(x, y, z);   // coors[:,3:0:-1]
    unsigned long long local = (unsigned long long)(i & (SEG - 1));
    k1[i] = ((unsigned long long)(unsigned)c1 << 16) | local;
    k2[i] = ((unsigned long long)(unsigned)c2 << 16) | local;
}

// ---------------------------------------------------------------- bitonic sort
__global__ void bitonic_stage_kernel(unsigned long long* __restrict__ k1,
                                     unsigned long long* __restrict__ k2,
                                     int k, int j) {
    int i = blockIdx.x * blockDim.x + threadIdx.x;
    int l = i ^ j;
    if (l <= i) return;
    bool asc = ((i & (SEG - 1) & k) == 0);   // direction from segment-local bit
    unsigned long long a = k1[i], b = k1[l];
    if ((a > b) == asc) { k1[i] = b; k1[l] = a; }
    a = k2[i]; b = k2[l];
    if ((a > b) == asc) { k2[i] = b; k2[l] = a; }
}

__global__ void extract_kernel(const unsigned long long* __restrict__ k1,
                               const unsigned long long* __restrict__ k2,
                               int* __restrict__ ind1, int* __restrict__ ind2,
                               int* __restrict__ inv1, int n) {
    int i = blockIdx.x * blockDim.x + threadIdx.x;
    if (i >= n) return;
    int base = i & ~(SEG - 1);
    int a = (int)(k1[i] & 0xFFFFull) + base;
    int b = (int)(k2[i] & 0xFFFFull) + base;
    ind1[i] = a;
    ind2[i] = b;
    inv1[a] = i;
}

__global__ void compose_kernel(const int* __restrict__ ind1, const int* __restrict__ ind2,
                               const int* __restrict__ inv1,
                               int* __restrict__ ind12, int* __restrict__ idxF, int n) {
    int i = blockIdx.x * blockDim.x + threadIdx.x;
    if (i >= n) return;
    int a = inv1[ind2[i]];
    ind12[i] = a;
    idxF[i]  = ind1[a];
}

// ---------------------------------------------------------------- weight prep
// dst[c * rows + r] = bf16(src[r * cols + c])  (transposed, n-major for WMMA B)
__global__ void transpose_bf16_kernel(const float* __restrict__ src,
                                      __bf16* __restrict__ dst, int rows, int cols) {
    int i = blockIdx.x * blockDim.x + threadIdx.x;
    if (i >= rows * cols) return;
    int r = i / cols, c = i - r * cols;
    dst[c * rows + r] = (__bf16)src[i];
}

// ---------------------------------------------------------------- WMMA helpers
// D(16xN) += A(16xKD) * B(KDxN) for one wave, one 16-col tile.
// Arow: bf16 row-major, pre-offset to this wave's row block, leading dim lda.
// Bt:   bf16 "n-major" (Bt[n][k]), pre-offset to col tile (nt*16 rows), leading dim ldb.
template<int KD>
__device__ inline floatx8 wave_gemm(const __bf16* Arow, int lda,
                                    const __bf16* Bt, int ldb,
                                    int lane, floatx8 acc) {
    const int m  = lane & 15;
    const int hi = lane >> 4;
#pragma unroll
    for (int kt = 0; kt < KD / 32; ++kt) {
        bf16x16 a, b;
        const __bf16* ap = Arow + m * lda + kt * 32 + hi * 8;
#pragma unroll
        for (int h = 0; h < 8; ++h) { a[h] = ap[h]; a[8 + h] = ap[16 + h]; }
        const __bf16* bp = Bt + (lane & 15) * ldb + kt * 32 + hi * 8;
#pragma unroll
        for (int h = 0; h < 8; ++h) { b[h] = bp[h]; b[8 + h] = bp[16 + h]; }
        acc = __builtin_amdgcn_wmma_f32_16x16x32_bf16(false, a, false, b,
                                                      (short)0, acc, false, false);
    }
    return acc;
}

__device__ inline void store_row_f32(float* dst, int ld, int lane, floatx8 d) {
    int col = lane & 15, rb = (lane >> 4) * 8;
#pragma unroll
    for (int v = 0; v < 8; ++v) dst[(rb + v) * ld + col] = d[v];
}

__device__ inline void store_row_bf16(__bf16* dst, int ld, int lane, floatx8 d, bool relu) {
    int col = lane & 15, rb = (lane >> 4) * 8;
#pragma unroll
    for (int v = 0; v < 8; ++v) {
        float x = d[v];
        if (relu) x = fmaxf(x, 0.f);
        dst[(rb + v) * ld + col] = (__bf16)x;
    }
}

__device__ inline void store_col_bf16(__bf16* dst, int ld, int lane, floatx8 d) {
    int col = lane & 15, rb = (lane >> 4) * 8;
#pragma unroll
    for (int v = 0; v < 8; ++v) dst[col * ld + rb + v] = (__bf16)d[v];
}

// ---------------------------------------------------------------- attention block
// One workgroup (4 waves / 128 threads) per group of 64 points.
template<int C, int COUT>
__global__ __launch_bounds__(128) void block_kernel(
    const int* __restrict__ idxFeat, const int* __restrict__ idxPts,
    const float* __restrict__ pts,  const float* __restrict__ feats,
    const float* __restrict__ Wpos, const float* __restrict__ bpos,
    const __bf16* __restrict__ WqT, const __bf16* __restrict__ WkT,
    const __bf16* __restrict__ WvT, const __bf16* __restrict__ WoT,
    const __bf16* __restrict__ WoutT, float* __restrict__ outp) {

    constexpr int RSZ  = GS * C * 2;                              // bf16 act buffer
    constexpr int R0SZ = (GS * GS * 4 > RSZ) ? GS * GS * 4 : RSZ; // h then S (f32)
    __shared__ __align__(16) char smem[R0SZ + 3 * RSZ];

    __bf16* sm_h  = (__bf16*)smem;                    // h (64xC)
    float*  sm_s  = (float*)smem;                     // scores (64x64), reuses h
    __bf16* sm_q  = (__bf16*)(smem + R0SZ);           // Q (64xC)
    __bf16* sm_p  = sm_q;                             // softmax P (64x64), reuses Q
    __bf16* sm_k  = (__bf16*)(smem + R0SZ + RSZ);     // K (64xC)
    __bf16* sm_o  = sm_k;                             // attn out (64xC), reuses K
    __bf16* sm_vt = (__bf16*)(smem + R0SZ + 2 * RSZ); // V^T (Cx64)
    __bf16* sm_r  = sm_vt;                            // relu(O@Wo) (64xC), reuses V^T
    float*  p3    = (float*)sm_vt;                    // staged points (only pre-QKV)
    float*  pmean = p3 + GS * 3;

    const int g = blockIdx.x, base = g * GS;
    const int tid = threadIdx.x, lane = tid & 31, w = tid >> 5;

    if (tid < GS) {
        int s = idxPts[base + tid];
        p3[tid * 3 + 0] = pts[s * 3 + 0];
        p3[tid * 3 + 1] = pts[s * 3 + 1];
        p3[tid * 3 + 2] = pts[s * 3 + 2];
    }
    __syncthreads();
    if (tid < 3) {
        float s = 0.f;
        for (int m = 0; m < GS; ++m) s += p3[m * 3 + tid];
        pmean[tid] = s * (1.f / GS);
    }
    __syncthreads();

    // h = x * ((p - mean) @ Wpos + bpos)   (bf16, row-major)
    for (int e = tid; e < GS * C; e += 128) {
        int m = e / C, c = e - m * C;
        int s = idxFeat[base + m];
        float pe = bpos[c];
#pragma unroll
        for (int d = 0; d < 3; ++d) pe += (p3[m * 3 + d] - pmean[d]) * Wpos[d * C + c];
        sm_h[e] = (__bf16)(feats[(size_t)s * C + c] * pe);
    }
    __syncthreads();

    const int row0 = w * 16;

    // Q = hWq, K = hWk, V = hWv  (V stored transposed)
    for (int nt = 0; nt < C / 16; ++nt) {
        floatx8 aq = {}, ak = {}, av = {};
        aq = wave_gemm<C>(sm_h + row0 * C, C, WqT + nt * 16 * C, C, lane, aq);
        ak = wave_gemm<C>(sm_h + row0 * C, C, WkT + nt * 16 * C, C, lane, ak);
        av = wave_gemm<C>(sm_h + row0 * C, C, WvT + nt * 16 * C, C, lane, av);
        store_row_bf16(sm_q + row0 * C + nt * 16, C, lane, aq, false);
        store_row_bf16(sm_k + row0 * C + nt * 16, C, lane, ak, false);
        store_col_bf16(sm_vt + nt * 16 * GS + row0, GS, lane, av);
    }
    __syncthreads();

    // S = Q K^T / sqrt(C)   (f32, overlays h)
    const float scale = rsqrtf((float)C);
    for (int nt = 0; nt < GS / 16; ++nt) {
        floatx8 s = {};
        s = wave_gemm<C>(sm_q + row0 * C, C, sm_k + nt * 16 * C, C, lane, s);
        int col = lane & 15, rb = (lane >> 4) * 8;
#pragma unroll
        for (int v = 0; v < 8; ++v)
            sm_s[(row0 + rb + v) * GS + nt * 16 + col] = s[v] * scale;
    }
    __syncthreads();

    // softmax rows -> P (bf16, overlays Q)
    if (tid < GS) {
        float mx = -3.4e38f;
        for (int j = 0; j < GS; ++j) mx = fmaxf(mx, sm_s[tid * GS + j]);
        float sum = 0.f;
        for (int j = 0; j < GS; ++j) {
            float ev = __expf(sm_s[tid * GS + j] - mx);
            sm_s[tid * GS + j] = ev;
            sum += ev;
        }
        float inv = 1.f / sum;
        for (int j = 0; j < GS; ++j) sm_p[tid * GS + j] = (__bf16)(sm_s[tid * GS + j] * inv);
    }
    __syncthreads();

    // O = P V   (bf16, overlays K)
    for (int nt = 0; nt < C / 16; ++nt) {
        floatx8 o = {};
        o = wave_gemm<GS>(sm_p + row0 * GS, GS, sm_vt + nt * 16 * GS, GS, lane, o);
        store_row_bf16(sm_o + row0 * C + nt * 16, C, lane, o, false);
    }
    __syncthreads();

    // R = relu(O @ Wo)   (bf16, overlays V^T)
    for (int nt = 0; nt < C / 16; ++nt) {
        floatx8 t = {};
        t = wave_gemm<C>(sm_o + row0 * C, C, WoT + nt * 16 * C, C, lane, t);
        store_row_bf16(sm_r + row0 * C + nt * 16, C, lane, t, true);
    }
    __syncthreads();

    // out = R @ Wout  (f32 to global)
    float* og = outp + (size_t)base * COUT;
    for (int nt = 0; nt < COUT / 16; ++nt) {
        floatx8 f = {};
        f = wave_gemm<C>(sm_r + row0 * C, C, WoutT + nt * 16 * C, C, lane, f);
        store_row_f32(og + row0 * COUT + nt * 16, COUT, lane, f);
    }
}

// ---------------------------------------------------------------- finalize meta
__global__ void finalize_meta_kernel(const int* __restrict__ idxF,
                                     const float* __restrict__ pts,
                                     const int* __restrict__ coors,
                                     const int* __restrict__ numbs,
                                     float* __restrict__ ptsOut,
                                     int* __restrict__ coorsOut,
                                     int* __restrict__ numbsOut, int n) {
    int i = blockIdx.x * blockDim.x + threadIdx.x;
    if (i < n) {
        int s = idxF[i];
        ptsOut[i * 3 + 0] = pts[s * 3 + 0];
        ptsOut[i * 3 + 1] = pts[s * 3 + 1];
        ptsOut[i * 3 + 2] = pts[s * 3 + 2];
        coorsOut[i * 4 + 0] = coors[s * 4 + 0];
        coorsOut[i * 4 + 1] = coors[s * 4 + 1];
        coorsOut[i * 4 + 2] = coors[s * 4 + 2];
        coorsOut[i * 4 + 3] = coors[s * 4 + 3];
    }
    if (i < 4) numbsOut[i] = numbs[i];
}

// ---------------------------------------------------------------- launcher
extern "C" void kernel_launch(void* const* d_in, const int* in_sizes, int n_in,
                              void* d_out, int out_size, void* d_ws, size_t ws_size,
                              hipStream_t stream) {
    (void)in_sizes; (void)n_in; (void)out_size; (void)ws_size;
    const int N = NPTS;

    const int*   numbs = (const int*)d_in[0];
    const int*   coors = (const int*)d_in[1];
    const float* feats = (const float*)d_in[2];
    const float* pts   = (const float*)d_in[3];
    const float* Wpos1 = (const float*)d_in[4];
    const float* bpos1 = (const float*)d_in[5];
    const float* Wpos2 = (const float*)d_in[11];
    const float* bpos2 = (const float*)d_in[12];

    // workspace carve
    char* ws = (char*)d_ws;
    size_t off = 0;
    auto nxt = [&](size_t bytes) -> char* {
        char* p = ws + off;
        off += (bytes + 255) & ~(size_t)255;
        return p;
    };
    unsigned long long* key1 = (unsigned long long*)nxt((size_t)N * 8);
    unsigned long long* key2 = (unsigned long long*)nxt((size_t)N * 8);
    int* ind1  = (int*)nxt((size_t)N * 4);
    int* ind2  = (int*)nxt((size_t)N * 4);
    int* inv1  = (int*)nxt((size_t)N * 4);
    int* ind12 = (int*)nxt((size_t)N * 4);
    int* idxF  = (int*)nxt((size_t)N * 4);
    __bf16* WqT1   = (__bf16*)nxt(64 * 64 * 2);
    __bf16* WkT1   = (__bf16*)nxt(64 * 64 * 2);
    __bf16* WvT1   = (__bf16*)nxt(64 * 64 * 2);
    __bf16* WoT1   = (__bf16*)nxt(64 * 64 * 2);
    __bf16* WoutT1 = (__bf16*)nxt(64 * 128 * 2);
    __bf16* WqT2   = (__bf16*)nxt(128 * 128 * 2);
    __bf16* WkT2   = (__bf16*)nxt(128 * 128 * 2);
    __bf16* WvT2   = (__bf16*)nxt(128 * 128 * 2);
    __bf16* WoT2   = (__bf16*)nxt(128 * 128 * 2);
    __bf16* WoutT2 = (__bf16*)nxt(128 * 128 * 2);
    float* out1 = (float*)nxt((size_t)N * 128 * 4);

    const int TB = 256, GB = (N + TB - 1) / TB;

    // 1) hilbert keys
    build_keys_kernel<<<GB, TB, 0, stream>>>(coors, key1, key2, N);

    // 2) per-segment bitonic sort (stable via packed index)
    for (int k = 2; k <= SEG; k <<= 1)
        for (int j = k >> 1; j >= 1; j >>= 1)
            bitonic_stage_kernel<<<GB, TB, 0, stream>>>(key1, key2, k, j);

    // 3) permutations
    extract_kernel<<<GB, TB, 0, stream>>>(key1, key2, ind1, ind2, inv1, N);
    compose_kernel<<<GB, TB, 0, stream>>>(ind1, ind2, inv1, ind12, idxF, N);

    // 4) weight prep (fp32 -> bf16, transposed)
    auto T = [&](int src, __bf16* dst, int r, int c) {
        int tot = r * c;
        transpose_bf16_kernel<<<(tot + TB - 1) / TB, TB, 0, stream>>>(
            (const float*)d_in[src], dst, r, c);
    };
    T(6, WqT1, 64, 64);   T(7, WkT1, 64, 64);   T(8, WvT1, 64, 64);
    T(9, WoT1, 64, 64);   T(10, WoutT1, 64, 128);
    T(13, WqT2, 128, 128); T(14, WkT2, 128, 128); T(15, WvT2, 128, 128);
    T(16, WoT2, 128, 128); T(17, WoutT2, 128, 128);

    // 5) block 1: gather via ind1, C0=64 -> C1=128
    block_kernel<64, 128><<<N / GS, 128, 0, stream>>>(
        ind1, ind1, pts, feats, Wpos1, bpos1,
        WqT1, WkT1, WvT1, WoT1, WoutT1, out1);

    // 6) meta outputs (pts, coors via composed perm; numbs)
    float* featsOut = (float*)d_out;
    float* ptsOut   = (float*)d_out + (size_t)N * 128;
    int*   coorsOut = (int*)d_out + (size_t)N * 131;
    int*   numbsOut = (int*)d_out + (size_t)N * 135;
    finalize_meta_kernel<<<GB, TB, 0, stream>>>(idxF, pts, coors, numbs,
                                                ptsOut, coorsOut, numbsOut, N);

    // 7) block 2: feats via ind12 (into out1), pts via idxF, C1=128 -> C2=128
    block_kernel<128, 128><<<N / GS, 128, 0, stream>>>(
        ind12, idxF, pts, out1, Wpos2, bpos2,
        WqT2, WkT2, WvT2, WoT2, WoutT2, featsOut);
}
